// Re_Encoder_37649683317500
// MI455X (gfx1250) — matrix operand order repaired and checked by reading, not compiled, not run on hardware
//
#include <hip/hip_runtime.h>
#include <stdint.h>

// ---------------- CDNA5 WMMA vector types ----------------
typedef __attribute__((ext_vector_type(16))) __bf16 v16bf;
typedef __attribute__((ext_vector_type(8)))  __bf16 v8bf;
typedef __attribute__((ext_vector_type(8)))  float  v8f;
typedef __attribute__((ext_vector_type(4)))  float  v4f;

#define R_R    8
#define N_N    128
#define D_D    512
#define H_H    256
#define TOPK   64
#define NPAIR  (N_N * N_N)     // 16384
#define KTILES (D_D / 32)      // 16 WMMA K-steps
#define NTILES (H_H / 16)      // 16 WMMA N-tiles
#define AROW   528             // padded LDS row stride (bf16 elems) to dodge bank conflicts
#define MBLK   64              // pairs per o2o block (4 WMMA M-tiles -> 4x B reuse)

static __device__ __forceinline__ unsigned short f32_to_bf16(float f) {
  unsigned int u = __float_as_uint(f);
  u += 0x7FFFu + ((u >> 16) & 1u);   // round-to-nearest-even
  return (unsigned short)(u >> 16);
}

static __device__ __forceinline__ v16bf load_frag(const unsigned short* p0,
                                                  const unsigned short* p1) {
  v8bf lo = *(const v8bf*)p0;
  v8bf hi = *(const v8bf*)p1;
  return __builtin_shufflevector(lo, hi, 0,1,2,3,4,5,6,7,8,9,10,11,12,13,14,15);
}

// ---------------------------------------------------------------------------
// Pack w1a/w1b [D,H] f32 row-major into WMMA B-fragment layout, bf16:
// dst[((kt*NTILES + nt)*32 + lane)*16 + e], where
//   lane 0-15 : col = nt*16+lane,     K = kt*32 + e        (e = 0..15)
//   lane16-31 : col = nt*16+lane-16,  K = kt*32 + 16 + e
// ---------------------------------------------------------------------------
__global__ __launch_bounds__(256) void pack_w_kernel(const float* __restrict__ w1a,
                                                     const float* __restrict__ w1b,
                                                     unsigned short* __restrict__ pa,
                                                     unsigned short* __restrict__ pb) {
  int gid   = blockIdx.x * 256 + threadIdx.x;  // 0 .. 262143 (2 matrices)
  int which = gid >> 17;
  int idx   = gid & 131071;
  int e    = idx & 15;
  int lane = (idx >> 4) & 31;
  int nt   = (idx >> 9) & 15;
  int kt   = (idx >> 13) & 15;
  int k = kt * 32 + (lane >> 4) * 16 + e;
  int h = nt * 16 + (lane & 15);
  const float* src = which ? w1b : w1a;
  unsigned short v = f32_to_bf16(src[k * H_H + h]);
  (which ? pb : pa)[idx] = v;
}

// ---------------------------------------------------------------------------
// Fused: obj2obj tile build -> WMMA MLP1 -> logit reduce; also accumulates
// the R-averaged pair features (register-resident, 64 f32/thread).
// 1 block = 64 pairs (4 M-tiles), 16 waves; wave owns 1 N-tile x 4 M-tiles,
// so each B fragment load feeds 4 WMMAs (quarters L2 weight traffic).
// ---------------------------------------------------------------------------
__global__ __launch_bounds__(512)
void o2o_kernel(const float* __restrict__ obj,    // [R,N,D]
                const float* __restrict__ geo,    // [R,N,N,D]
                const float* __restrict__ maskv,  // [N*N]
                const unsigned short* __restrict__ w1p,  // packed bf16 B frags
                const float* __restrict__ b1,     // [H]
                const float* __restrict__ w2,     // [H] (w2a[:,0])
                const float* __restrict__ b2,     // [1]
                float* __restrict__ out_logits,   // d_out[0..NPAIR)
                float* __restrict__ probs,        // ws [NPAIR]
                float* __restrict__ afeat) {      // ws [NPAIR, D]
  __shared__ __align__(16) unsigned short Abuf[MBLK * AROW];  // 67,584 B
  __shared__ float Lacc[MBLK];

  const int tid  = threadIdx.x;
  const int wave = tid >> 5;       // 0..15 : this wave's N-tile
  const int lane = tid & 31;
  const int p0   = blockIdx.x * MBLK;

  if (tid < MBLK) Lacc[tid] = 0.f;

  // per-lane column constants for this wave's N-tile
  const int col = wave * 16 + (lane & 15);
  const float b1c = b1[col];
  const float w2c = w2[col];

  const int mrow   = tid >> 3;        // 0..63 : pair row this thread stages
  const int dchunk = (tid & 7) * 64;  // 64 contiguous d per thread
  const int pi = (p0 + mrow) / N_N;
  const int pj = (p0 + mrow) % N_N;

  const int arow  = lane & 15;
  const int alo   = (lane < 16) ? 0 : 16;  // A-frag K sub-slice per half-wave
  const int rbase = (lane < 16) ? 0 : 8;  // C rows: VGPR g -> M = rbase+g

  // register-resident feature-sum accumulator (thread owns fixed 64 elems)
  float sreg[64];
#pragma unroll
  for (int q = 0; q < 64; ++q) sreg[q] = 0.f;

  __syncthreads();

  for (int r = 0; r < R_R; ++r) {
    // ---- build bf16 obj2obj tile + fp32 feature sum (in regs) ----
    const float* oi = obj + ((size_t)r * N_N + pi) * D_D + dchunk;
    const float* oj = obj + ((size_t)r * N_N + pj) * D_D + dchunk;
    const float* gg = geo + (((size_t)r * N_N + pi) * N_N + pj) * D_D + dchunk;
    unsigned short* adst = Abuf + mrow * AROW + dchunk;
#pragma unroll
    for (int q = 0; q < 64; q += 4) {
      v4f a = *(const v4f*)(oi + q);
      v4f b = *(const v4f*)(oj + q);
      v4f g = *(const v4f*)(gg + q);
#pragma unroll
      for (int c = 0; c < 4; ++c) {
        float v = a[c] * b[c] + g[c];
        adst[q + c] = f32_to_bf16(v);
        sreg[q + c] += v;
      }
    }
    // prefetch next r's geo stream (global_prefetch_b8) to hide HBM latency
    if (r + 1 < R_R) {
      const char* nx =
          (const char*)(geo + (((size_t)(r + 1) * N_N + pi) * N_N + pj) * D_D + dchunk);
      __builtin_prefetch(nx, 0, 1);
      __builtin_prefetch(nx + 128, 0, 1);
    }
    __syncthreads();

    // ---- WMMA: [64 x 512] x [512 x 16(this wave)] -> [64 x 16] f32 ----
    v8f acc[4];
#pragma unroll
    for (int mt = 0; mt < 4; ++mt) acc[mt] = (v8f){};

#pragma unroll 1
    for (int kt = 0; kt < KTILES; ++kt) {
      const unsigned short* bs =
          w1p + (((size_t)kt * NTILES + wave) * 32 + lane) * 16;
      v16bf bfrag = load_frag(bs, bs + 8);
#pragma unroll
      for (int mt = 0; mt < 4; ++mt) {
        const unsigned short* ab = Abuf + (mt * 16 + arow) * AROW + kt * 32;
        v16bf afrag = load_frag(ab + alo, ab + alo + 8);
        acc[mt] = __builtin_amdgcn_wmma_f32_16x16x32_bf16(
            false, afrag, false, bfrag, (short)0, acc[mt], false, false);
      }
    }

    // ---- +b1, ReLU, dot with w2, reduce into per-pair logits ----
#pragma unroll
    for (int mt = 0; mt < 4; ++mt) {
#pragma unroll
      for (int g = 0; g < 8; ++g) {
        float h = acc[mt][g] + b1c;
        h = h > 0.f ? h : 0.f;
        atomicAdd(&Lacc[mt * 16 + rbase + g], h * w2c);
      }
    }
    __syncthreads();
  }

  // averaged pair features -> workspace
  {
    float* fd = afeat + (size_t)(p0 + mrow) * D_D + dchunk;
#pragma unroll
    for (int q = 0; q < 64; ++q) fd[q] = sreg[q] * (1.f / R_R);
  }
  if (tid < MBLK) {
    int p = p0 + tid;
    float lg = Lacc[tid] * (1.f / R_R) + b2[0];
    out_logits[p] = lg;
    float pr = 1.f / (1.f + __expf(-lg));
    probs[p] = pr * maskv[p];
  }
}

// ---------------------------------------------------------------------------
// Top-64 of 16384 probs; stable (lower index wins ties), like jax.lax.top_k.
// 1 block x 256 threads; candidates register-resident; 64-bit taken mask.
// ---------------------------------------------------------------------------
__global__ __launch_bounds__(256)
void topk_kernel(const float* __restrict__ probs,
                 int* __restrict__ inds_out,
                 float* __restrict__ inds_f_out) {
  __shared__ float s_bv[8];
  __shared__ int   s_bi[8];
  __shared__ int   s_win;
  __shared__ int   s_sel[TOPK];

  const int tid  = threadIdx.x;
  const int wave = tid >> 5;
  const int lane = tid & 31;

  float vals[64];
#pragma unroll
  for (int k = 0; k < 64; ++k) vals[k] = probs[k * 256 + tid];
  unsigned long long taken = 0ull;

#pragma unroll 1
  for (int t = 0; t < TOPK; ++t) {
    float bv = -3.4e38f;
    int   bi = 0x7FFFFFFF;
#pragma unroll
    for (int k = 0; k < 64; ++k) {
      if (!((taken >> k) & 1ull)) {
        float v = vals[k];
        int   i = (k << 8) | tid;
        if (v > bv || (v == bv && i < bi)) { bv = v; bi = i; }
      }
    }
    // wave32 shuffle reduce
#pragma unroll
    for (int off = 16; off > 0; off >>= 1) {
      float ov = __shfl_down(bv, (unsigned)off);
      int   oi = __shfl_down(bi, (unsigned)off);
      if (ov > bv || (ov == bv && oi < bi)) { bv = ov; bi = oi; }
    }
    if (lane == 0) { s_bv[wave] = bv; s_bi[wave] = bi; }
    __syncthreads();
    if (tid == 0) {
      float gv = s_bv[0]; int gi = s_bi[0];
#pragma unroll
      for (int w = 1; w < 8; ++w) {
        if (s_bv[w] > gv || (s_bv[w] == gv && s_bi[w] < gi)) { gv = s_bv[w]; gi = s_bi[w]; }
      }
      s_win = gi;
      s_sel[t] = gi;
    }
    __syncthreads();
    int win = s_win;
    if ((win & 255) == tid) taken |= 1ull << (win >> 8);
    __syncthreads();
  }

  if (tid < TOPK) {
    int i = s_sel[tid];
    inds_out[tid]   = i;
    inds_f_out[tid] = (float)i;
  }
}

// ---------------------------------------------------------------------------
// p2p: outer products of top-k averaged pair features -> MLP2 via WMMA.
// 1 block = 16 (a,b) pairs, 4 waves.
// ---------------------------------------------------------------------------
__global__ __launch_bounds__(128)
void p2p_kernel(const float* __restrict__ afeat,  // ws [NPAIR, D]
                const int* __restrict__ inds,     // ws [TOPK]
                const unsigned short* __restrict__ w1p,  // packed w1b
                const float* __restrict__ b1,     // [H]
                const float* __restrict__ w2,     // [H]
                const float* __restrict__ b2,     // [1]
                float* __restrict__ out) {        // d_out + NPAIR, [k*k]
  __shared__ __align__(16) unsigned short Abuf[16 * AROW];
  __shared__ float Lacc[16];

  const int tid  = threadIdx.x;
  const int wave = tid >> 5;
  const int lane = tid & 31;
  const int pp0  = blockIdx.x * 16;

  if (tid < 16) Lacc[tid] = 0.f;

  float b1c[4], w2c[4];
#pragma unroll
  for (int t = 0; t < 4; ++t) {
    int col = (wave * 4 + t) * 16 + (lane & 15);
    b1c[t] = b1[col];
    w2c[t] = w2[col];
  }

  const int mrow   = tid >> 3;
  const int dchunk = (tid & 7) * 64;
  const int pp = pp0 + mrow;
  const int ia = inds[pp >> 6];   // row index a
  const int ib = inds[pp & 63];   // row index b

  // build bf16 p2p tile
  {
    const float* fa = afeat + (size_t)ia * D_D + dchunk;
    const float* fb = afeat + (size_t)ib * D_D + dchunk;
    unsigned short* adst = Abuf + mrow * AROW + dchunk;
#pragma unroll
    for (int q = 0; q < 64; q += 4) {
      v4f a = *(const v4f*)(fa + q);
      v4f b = *(const v4f*)(fb + q);
#pragma unroll
      for (int c = 0; c < 4; ++c) adst[q + c] = f32_to_bf16(a[c] * b[c]);
    }
  }
  __syncthreads();

  const int arow  = lane & 15;
  const int alo   = (lane < 16) ? 0 : 16;
  const int rbase = (lane < 16) ? 0 : 8;

  v8f acc[4];
#pragma unroll
  for (int t = 0; t < 4; ++t) acc[t] = (v8f){};

#pragma unroll 1
  for (int kt = 0; kt < KTILES; ++kt) {
    const unsigned short* ab = Abuf + arow * AROW + kt * 32;
    v16bf afrag = load_frag(ab + alo, ab + alo + 8);
#pragma unroll
    for (int t = 0; t < 4; ++t) {
      const unsigned short* bs =
          w1p + (((size_t)kt * NTILES + (wave * 4 + t)) * 32 + lane) * 16;
      v16bf bfrag = load_frag(bs, bs + 8);
      acc[t] = __builtin_amdgcn_wmma_f32_16x16x32_bf16(
          false, afrag, false, bfrag, (short)0, acc[t], false, false);
    }
  }

  float part[8];
#pragma unroll
  for (int g = 0; g < 8; ++g) part[g] = 0.f;
#pragma unroll
  for (int t = 0; t < 4; ++t) {
#pragma unroll
    for (int g = 0; g < 8; ++g) {
      float h = acc[t][g] + b1c[t];
      h = h > 0.f ? h : 0.f;
      part[g] += h * w2c[t];
    }
  }
#pragma unroll
  for (int g = 0; g < 8; ++g) atomicAdd(&Lacc[rbase + g], part[g]);
  __syncthreads();

  if (tid < 16) out[pp0 + tid] = Lacc[tid] + b2[0];
}

// ---------------------------------------------------------------------------
extern "C" void kernel_launch(void* const* d_in, const int* in_sizes, int n_in,
                              void* d_out, int out_size, void* d_ws, size_t ws_size,
                              hipStream_t stream) {
  (void)in_sizes; (void)n_in; (void)out_size; (void)ws_size;

  const float* obj   = (const float*)d_in[0];
  const float* geo   = (const float*)d_in[1];
  const float* maskv = (const float*)d_in[2];
  const float* w1a   = (const float*)d_in[3];
  const float* b1a   = (const float*)d_in[4];
  const float* w2a   = (const float*)d_in[5];
  const float* b2a   = (const float*)d_in[6];
  const float* w1b   = (const float*)d_in[7];
  const float* b1b   = (const float*)d_in[8];
  const float* w2b   = (const float*)d_in[9];
  const float* b2b   = (const float*)d_in[10];
  float* out = (float*)d_out;

  // workspace layout (bytes)
  char* ws = (char*)d_ws;
  float* afeat = (float*)(ws);                       // 16384*512*4 = 33,554,432
  float* probs = (float*)(ws + 33554432u);           // 16384*4     =     65,536
  int*   inds  = (int*)  (ws + 33619968u);           // 64*4  (padded to 256)
  unsigned short* w1ap = (unsigned short*)(ws + 33620224u);  // 262,144
  unsigned short* w1bp = (unsigned short*)(ws + 33882368u);  // 262,144

  pack_w_kernel<<<1024, 256, 0, stream>>>(w1a, w1b, w1ap, w1bp);
  o2o_kernel<<<NPAIR / MBLK, 512, 0, stream>>>(obj, geo, maskv, w1ap, b1a, w2a,
                                               b2a, out, probs, afeat);
  topk_kernel<<<1, 256, 0, stream>>>(probs, inds, out + NPAIR + TOPK * TOPK);
  p2p_kernel<<<(TOPK * TOPK) / 16, 128, 0, stream>>>(afeat, inds, w1bp, b1b,
                                                     w2b, b2b, out + NPAIR);
}
